// CrossEntropy_55568286876202
// MI455X (gfx1250) — matrix operand order repaired and checked by reading, not compiled, not run on hardware
//
#include <hip/hip_runtime.h>
#include <stdint.h>

// Problem constants (B=2, T=2048, D=1024, V=50257)
#define D_   1024
#define V_   50257
#define NT_  4096                      // B*T tokens
#define NB_  128                       // vocab columns per workgroup (slice)
#define NS_  ((V_ + NB_ - 1) / NB_)    // 393 slices
#define KCH_ (D_ / 32)                 // 32 K-chunks of 32 (bf16 WMMA K)

typedef __attribute__((ext_vector_type(16))) __bf16 v16bf;
typedef __attribute__((ext_vector_type(8)))  float  v8f;

union BF16x16 {
  v16bf v;
  unsigned short h[16];
  uint4 q[2];
};

// fp32 -> bf16 round-to-nearest-even (bit trick; NaN edge ignored)
__device__ __forceinline__ unsigned short f2bf(float f) {
  unsigned u = __float_as_uint(f);
  u += 0x7fffu + ((u >> 16) & 1u);
  return (unsigned short)(u >> 16);
}

__global__ void ce_zero(float* __restrict__ out) { out[0] = 0.0f; }

// ---------------------------------------------------------------------------
// Kernel 0: convert hidden activations fp32 -> bf16 ONCE, stored in WMMA
// A-fragment order:  Apk[((mt*KCH + c)*32 + lane)*8 + idx/2]  (2 bf16 / u32).
// A-frag element i (0..15): K = 32c + (i/8)*16 + (lane>>4)*8 + (i&7), M = lane&15.
// Hot loop then does 2x global_load_b128 per chunk with zero conversion VALU.
// ---------------------------------------------------------------------------
__global__ __launch_bounds__(256) void ce_cvtA(const float* __restrict__ hid,
                                               uint32_t* __restrict__ Apk)
{
  const int e = blockIdx.x * 256 + threadIdx.x;   // one float4 per thread
  const int token = e >> 8;                       // D_/4 = 256 float4 per token
  const int k4    = (e & 255) << 2;               // k base (multiple of 4)
  const float4 f  = ((const float4*)hid)[e];      // row-major (token,k) == flat e

  const int mt = token >> 4, m = token & 15;
  const int c    = k4 >> 5;
  const int r    = k4 & 31;
  const int half = (r >> 3) & 1;
  const int g    = r >> 4;
  const int j    = r & 7;                         // 0 or 4
  const int l    = (half << 4) | m;
  const int idx  = g * 8 + j;
  const size_t slot = ((size_t)(mt * KCH_ + c) * 32 + l) * 8 + (idx >> 1);

  uint2 p;
  p.x = (uint32_t)f2bf(f.x) | ((uint32_t)f2bf(f.y) << 16);
  p.y = (uint32_t)f2bf(f.z) | ((uint32_t)f2bf(f.w) << 16);
  *(uint2*)(Apk + slot) = p;                      // slot is even -> 8B aligned
}

// ---------------------------------------------------------------------------
// Kernel 1: one block per 128-wide vocab slice.
//   Stage:  E-slice (128 x 1024 fp32) -> bf16 WMMA-B fragments in 256KB LDS.
//   Stream: all 4096 tokens in 16-row tiles (8 waves round-robin); per K-chunk
//           load A frag (2x b128, pre-converted) + all 8 B frags (16x ds b128),
//           then 8 back-to-back v_wmma_f32_16x16x32_bf16.
// ---------------------------------------------------------------------------
__global__ __launch_bounds__(256) void ce_partial(
    const uint32_t* __restrict__ Apk,   // pre-converted A fragments
    const float* __restrict__ emb,      // [V_, D_]
    const float* __restrict__ bias,     // [V_]
    const int*   __restrict__ labels,   // [NT_] (harness contract: integer -> int*)
    float* __restrict__ maxP,           // [NS_, NT_]
    float* __restrict__ sumP,           // [NS_, NT_]
    float* __restrict__ labP)           // [NS_, NT_]
{
  extern __shared__ uint32_t lds[];               // 256KB B-frags + 512B bias
  uint32_t* Bpk  = lds;                           // NB_*D_/2 uint32 (2 bf16 each)
  float*    bLds = (float*)(lds + NB_ * D_ / 2);  // NB_ floats

  const int slice = blockIdx.x;
  const int n0    = slice * NB_;
  const int tid   = threadIdx.x;
  const int lane  = tid & 31;
  const int wave  = tid >> 5;
  const int half  = lane >> 4;    // which 16-lane half
  const int lx    = lane & 15;

  // ---- Stage E-slice into LDS as pre-swizzled bf16 B-fragments ----
  // B frag (chunk c, subtile s, lane l) elem i = E[n0+16s+(l&15)][32c + (i/8)*16 + (l>>4)*8 + (i&7)]
  for (int e = tid; e < NB_ * D_ / 4; e += 256) {
    const int j  = e >> 8;               // column within slice (D_/4 = 256 f4/col)
    const int k4 = (e & 255) << 2;       // k base (multiple of 4)
    int row = n0 + j; if (row >= V_) row = V_ - 1;   // clamp OOB cols
    const float4 f = ((const float4*)(emb + (size_t)row * D_))[k4 >> 2];
    const int s = j >> 4, lanelo = j & 15;
    const int c  = k4 >> 5;
    const int hk = (k4 >> 3) & 1;
    const int l  = (hk << 4) | lanelo;
    const int slot = (((c * 8 + s) * 32 + l) << 3) + (((((k4 >> 4) & 1) << 3) + (k4 & 7)) >> 1);
    Bpk[slot]     = (uint32_t)f2bf(f.x) | ((uint32_t)f2bf(f.y) << 16);
    Bpk[slot + 1] = (uint32_t)f2bf(f.z) | ((uint32_t)f2bf(f.w) << 16);
  }
  if (tid < NB_) {
    const int n = n0 + tid;
    bLds[tid] = (n < V_) ? bias[n] : -1e30f;       // OOB col -> ~-inf logit
  }
  __syncthreads();

  // ---- Stream tokens: wave w handles m-tiles w, w+8, ... ----
  for (int mt = wave; mt < NT_ / 16; mt += 8) {
    const int tok0 = mt * 16;
    const uint32_t* apk = Apk + ((size_t)mt * (KCH_ * 32) + lane) * 8;
    v8f acc[8] = {};

    for (int c = 0; c < KCH_; ++c) {
      // A fragment: 32B, pre-converted & pre-swizzled (fully coalesced)
      const uint4* ap = (const uint4*)(apk + (size_t)c * 256);
      BF16x16 A;
      A.q[0] = ap[0];
      A.q[1] = ap[1];

      // All 8 B fragments of this chunk up front (batched ds_load_b128)
      BF16x16 Bf[8];
#pragma unroll
      for (int s = 0; s < 8; ++s) {
        const uint4* bp = (const uint4*)(Bpk + (((c * 8 + s) * 32 + lane) << 3));
        Bf[s].q[0] = bp[0];
        Bf[s].q[1] = bp[1];
      }
#pragma unroll
      for (int s = 0; s < 8; ++s) {
        acc[s] = __builtin_amdgcn_wmma_f32_16x16x32_bf16(
            /*neg_a=*/false, A.v, /*neg_b=*/false, Bf[s].v,
            /*c_mod=*/(short)0, acc[s], /*reuse_a=*/false, /*reuse_b=*/false);
      }
    }

    // ---- Epilogue: bias + slice-local online-softmax stats per token ----
#pragma unroll
    for (int j = 0; j < 8; ++j) {
      const int row   = j + 8 * half;          // C layout: VGPR j -> M = j + 8*half
      const int token = tok0 + row;
      const int lab   = labels[token];
      float lg[8];
#pragma unroll
      for (int s = 0; s < 8; ++s) lg[s] = acc[s][j] + bLds[s * 16 + lx];

      float lm = lg[0];
#pragma unroll
      for (int s = 1; s < 8; ++s) lm = fmaxf(lm, lg[s]);
#pragma unroll
      for (int off = 1; off < 16; off <<= 1)   // stays within 16-lane half
        lm = fmaxf(lm, __shfl_xor(lm, off, 32));

      float ls = 0.0f, lb = 0.0f;
      const int dd = lab - n0 - lx;            // matches col n0 + 16s + lx
#pragma unroll
      for (int s = 0; s < 8; ++s) {
        ls += __expf(lg[s] - lm);
        lb += (dd == s * 16) ? lg[s] : 0.0f;
      }
#pragma unroll
      for (int off = 1; off < 16; off <<= 1) {
        ls += __shfl_xor(ls, off, 32);
        lb += __shfl_xor(lb, off, 32);
      }

      if (lx == 0) {                           // lanes 0 and 16 write their rows
        const size_t o = (size_t)slice * NT_ + token;
        maxP[o] = lm;
        sumP[o] = ls;
        labP[o] = lb;
      }
    }
  }
}

// ---------------------------------------------------------------------------
// Kernel 2: one block per token; merge 393 slice partials (online-softmax
// combine), then atomically accumulate the mean NLL.
// ---------------------------------------------------------------------------
__global__ __launch_bounds__(64) void ce_reduce(
    const float* __restrict__ maxP, const float* __restrict__ sumP,
    const float* __restrict__ labP, float* __restrict__ out)
{
  const int token = blockIdx.x;
  const int tid   = threadIdx.x;
  float m = -1e30f, s = 0.0f, lb = 0.0f;
  for (int sl = tid; sl < NS_; sl += 64) {
    const size_t o = (size_t)sl * NT_ + token;
    const float m2 = maxP[o], s2 = sumP[o];
    const float nm = fmaxf(m, m2);
    s = s * __expf(m - nm) + s2 * __expf(m2 - nm);
    m = nm;
    lb += labP[o];
  }
  __shared__ float sm[64], ss[64], sb[64];
  sm[tid] = m; ss[tid] = s; sb[tid] = lb;
  __syncthreads();
  for (int w = 32; w > 0; w >>= 1) {
    if (tid < w) {
      const float m1 = sm[tid], s1 = ss[tid];
      const float m2 = sm[tid + w], s2 = ss[tid + w];
      const float nm = fmaxf(m1, m2);
      sm[tid] = nm;
      ss[tid] = s1 * __expf(m1 - nm) + s2 * __expf(m2 - nm);
      sb[tid] += sb[tid + w];
    }
    __syncthreads();
  }
  if (tid == 0) {
    const float loss = (__logf(ss[0]) + sm[0]) - sb[0];   // logsumexp - label logit
    atomicAdd(out, loss * (1.0f / (float)NT_));
  }
}

// ---------------------------------------------------------------------------
extern "C" void kernel_launch(void* const* d_in, const int* in_sizes, int n_in,
                              void* d_out, int out_size, void* d_ws, size_t ws_size,
                              hipStream_t stream) {
  (void)in_sizes; (void)n_in; (void)out_size; (void)ws_size;
  const float* hid    = (const float*)d_in[0];   // [B,T,D] fp32
  const float* emb    = (const float*)d_in[1];   // [V,D]   fp32
  const float* bias   = (const float*)d_in[2];   // [V]     fp32
  const int*   labels = (const int*)d_in[3];     // [B,T]   (harness: integer -> int*)
  float* out = (float*)d_out;

  // d_ws layout: partials (3 * NS_*NT_ fp32 ~= 19.3MB) + packed A (8.4MB)
  float* maxP = (float*)d_ws;
  float* sumP = maxP + (size_t)NS_ * NT_;
  float* labP = sumP + (size_t)NS_ * NT_;
  uint32_t* Apk = (uint32_t*)(labP + (size_t)NS_ * NT_);

  const int shmem = NB_ * D_ / 2 * 4 + NB_ * 4;  // 256KB frags + bias (<= 320KB/WGP)
  hipFuncSetAttribute(reinterpret_cast<const void*>(ce_partial),
                      hipFuncAttributeMaxDynamicSharedMemorySize, shmem);

  ce_cvtA<<<NT_ * D_ / 4 / 256, 256, 0, stream>>>(hid, Apk);
  ce_zero<<<1, 1, 0, stream>>>(out);
  ce_partial<<<NS_, 256, shmem, stream>>>(Apk, emb, bias, labels, maxP, sumP, labP);
  ce_reduce<<<NT_, 64, 0, stream>>>(maxP, sumP, labP, out);
}
